// GraphDiffNetSequential_46720654246028
// MI455X (gfx1250) — matrix operand order repaired and checked by reading, not compiled, not run on hardware
//
#include <hip/hip_runtime.h>
#include <hip/hip_bf16.h>

// ---------------------------------------------------------------------------
// GraphDiffNet forward for MI455X (gfx1250, wave32, WMMA).
// All GEMMs have N=64 -> one specialized bf16-WMMA GEMM kernel (fp32 in/out,
// fp32 accumulate, bf16 convert in the LDS tile loader). gradX/gradY (128MB)
// and rbf/rbfT (64MB) stay L2-resident across the 4 blocks (192MB L2).
// Round 3: double-buffered LDS tiles (ping-pong), one barrier per K-step,
// next tile's global loads issued before the current step's WMMAs.
// ---------------------------------------------------------------------------

static constexpr int kNV = 4096;
static constexpr int kNG = 2048;
static constexpr int kD  = 64;
static constexpr int kK  = 128;
static constexpr int kNB = 4;
static constexpr int kCOUT = 8;
static constexpr int kE  = 32768;

typedef __bf16 bf16_t;
typedef bf16_t v16bf __attribute__((ext_vector_type(16)));
typedef float  v8f   __attribute__((ext_vector_type(8)));
typedef unsigned int u32;

static inline int cdiv(int a, int b) { return (a + b - 1) / b; }

// ---------------------------------------------------------------------------
// WMMA GEMM: C[M,64] = epilogue(alpha * A[M,K] @ B[K,64]).
// 128 threads = 4 waves; block computes a 64x64 tile, each wave a 16x64 strip
// as 4 v_wmma_f32_16x16x32_bf16 tiles; K in steps of 32, double-buffered.
// Requires M % 64 == 0, K % 32 == 0 (true for all call sites here).
// ---------------------------------------------------------------------------
__global__ __launch_bounds__(128) void gemm_wmma_n64(
    const float* __restrict__ A, int lda,
    const float* __restrict__ B, int ldb,
    float* __restrict__ C,
    const float* __restrict__ bias,    // nullable, [64]
    const float* __restrict__ addbuf,  // nullable, [M,64] residual add
    int M, int Kdim, float alpha, int accumulate, int do_relu)
{
  // Row stride 40 elems = 80B = 20 dwords: 16B-aligned every row (80 = 16*5)
  // and bank-conflict-free for 16-lane row fans (gcd(20,64)=4 -> cycle 16).
  __shared__ __align__(16) bf16_t As[2][64][40];  // (m,k) tiles, 64x32 used
  __shared__ __align__(16) bf16_t Bt[2][64][40];  // (n,k) = B^T tiles

  const int tid  = threadIdx.x;
  const int lane = tid & 31;
  const int wave = tid >> 5;
  const int half = lane >> 4;   // 0: lanes 0-15, 1: lanes 16-31
  const int mn   = lane & 15;
  const int m0   = blockIdx.x * 64;

  // A fill mapping: thread -> rows {a_r0 + 16i}, columns [a_c, a_c+4)
  const int a_r0 = tid >> 3;        // 0..15
  const int a_c  = (tid & 7) * 4;   // 0,4,...,28
  // B fill mapping: thread -> column n, K rows [b_k0, b_k0+16)
  const int b_n  = tid & 63;
  const int b_k0 = (tid >> 6) * 16; // 0 or 16

  float4 av[4];
  float  bv[16];

  // Stage one K-tile's global loads into registers.
  auto load_tile = [&](int k0) {
    #pragma unroll
    for (int i = 0; i < 4; ++i)
      av[i] = *(const float4*)&A[(size_t)(m0 + a_r0 + 16 * i) * lda + (k0 + a_c)];
    #pragma unroll
    for (int j = 0; j < 16; ++j)
      bv[j] = B[(size_t)(k0 + b_k0 + j) * ldb + b_n];
  };
  // Convert staged registers to bf16 and store into LDS buffer `buf`.
  auto store_tile = [&](int buf) {
    #pragma unroll
    for (int i = 0; i < 4; ++i) {
      union { bf16_t h[4]; uint2 u; } pk;
      pk.h[0] = (bf16_t)av[i].x; pk.h[1] = (bf16_t)av[i].y;
      pk.h[2] = (bf16_t)av[i].z; pk.h[3] = (bf16_t)av[i].w;
      *(uint2*)&As[buf][a_r0 + 16 * i][a_c] = pk.u;   // 8B store, 8B-aligned
    }
    union { bf16_t h[16]; uint4 u[2]; } pk;
    #pragma unroll
    for (int j = 0; j < 16; ++j) pk.h[j] = (bf16_t)bv[j];
    uint4* dst = (uint4*)&Bt[buf][b_n][b_k0];         // 16B-aligned
    dst[0] = pk.u[0];
    dst[1] = pk.u[1];
  };

  v8f acc[4] = {};

  load_tile(0);
  store_tile(0);

  int it = 0;
  for (int k0 = 0; k0 < Kdim; k0 += 32, it ^= 1) {
    __syncthreads();  // buf[it] ready; prev reads of buf[it^1] complete
    const bool more = (k0 + 32) < Kdim;
    if (more) load_tile(k0 + 32);   // overlap with frag build + WMMA below

    union Frag { v16bf v; u32 u[8]; };

    // A fragment, ISA 16-bit A 16x32 layout:
    //   half=0: e<8 -> K=e,   e>=8 -> K=8+e  ; half=1: +8 / +8
    Frag a;
    const u32* arow = (const u32*)(&As[it][wave * 16 + mn][0]);
    #pragma unroll
    for (int j = 0; j < 4; ++j) a.u[j]     = arow[4 * half + j];       // K = 8h + 2j
    #pragma unroll
    for (int j = 0; j < 4; ++j) a.u[4 + j] = arow[8 + 4 * half + j];   // K = 16 + 8h + 2j

    #pragma unroll
    for (int t = 0; t < 4; ++t) {
      // B fragment, 32x16: lane n = mn, same K order as A (e<8 -> K=8h+e, e>=8 -> K=16+8h+e-8)
      Frag b;
      const u32* brow = (const u32*)(&Bt[it][t * 16 + mn][0]);
      #pragma unroll
      for (int j = 0; j < 8; ++j) b.u[j] = brow[j < 4 ? 4 * half + j : 8 + 4 * half + (j - 4)];
      acc[t] = __builtin_amdgcn_wmma_f32_16x16x32_bf16(
          false, a.v, false, b.v, (short)0, acc[t], false, false);
    }

    if (more) store_tile(it ^ 1);   // safe: buf[it^1] reads finished pre-barrier
  }

  // Epilogue per ISA C/D layout: VGPR r, lane -> (M = r + 8*half, N = mn)
  #pragma unroll
  for (int t = 0; t < 4; ++t) {
    int n = t * 16 + mn;
    float bvv = bias ? bias[n] : 0.0f;
    #pragma unroll
    for (int r = 0; r < 8; ++r) {
      int m = m0 + wave * 16 + half * 8 + r;
      size_t off = (size_t)m * 64 + n;
      float v = alpha * acc[t][r];
      if (accumulate) v += C[off];
      v += bvv;
      if (addbuf) v += addbuf[off];
      if (do_relu) v = fmaxf(v, 0.0f);
      C[off] = v;
    }
  }
}

// ---------------------------------------------------------------------------
// Small elementwise / reduction / scatter kernels
// ---------------------------------------------------------------------------
__global__ void k_rbf(const float* __restrict__ vert, const float* __restrict__ gpos,
                      float* __restrict__ rbf, float* __restrict__ rbfT) {
  int idx = blockIdx.x * blockDim.x + threadIdx.x;
  if (idx >= kNV * kNG) return;
  int v = idx >> 11, g = idx & (kNG - 1);
  float dx = vert[v * 3 + 0] - gpos[g * 3 + 0];
  float dy = vert[v * 3 + 1] - gpos[g * 3 + 1];
  float dz = vert[v * 3 + 2] - gpos[g * 3 + 2];
  float dist = sqrtf(fmaxf(dx * dx + dy * dy + dz * dz, 0.0f));
  float r = __expf(-dist * (1.0f / 2.5f));
  rbf[(size_t)v * kNG + g] = r;
  rbfT[(size_t)g * kNV + v] = r;
}

__global__ void k_lin1(const float* __restrict__ sx, const float* __restrict__ w,
                       const float* __restrict__ b, float* __restrict__ out) {
  int idx = blockIdx.x * blockDim.x + threadIdx.x;
  if (idx >= kNV * kD) return;
  int v = idx >> 6, d = idx & 63;
  float acc = b[d];
  #pragma unroll
  for (int j = 0; j < 5; ++j) acc += sx[v * 5 + j] * w[j * kD + d];
  out[idx] = acc;
}

__global__ void k_xm(const float* __restrict__ x, const float* __restrict__ mass,
                     float* __restrict__ xm) {
  int idx = blockIdx.x * blockDim.x + threadIdx.x;
  if (idx >= kNV * kD) return;
  xm[idx] = x[idx] * mass[idx >> 6];
}

// x_spec[k][d] = sum_v evecs[v][k] * xm[v][d];  grid = K blocks, block = D threads
__global__ void k_spec(const float* __restrict__ evecs, const float* __restrict__ xm,
                       float* __restrict__ xspec) {
  int k = blockIdx.x, d = threadIdx.x;
  float acc = 0.0f;
  for (int v = 0; v < kNV; ++v)
    acc += evecs[(size_t)v * kK + k] * xm[(size_t)v * kD + d];
  xspec[k * kD + d] = acc;
}

__global__ void k_spec2(const float* __restrict__ evals, const float* __restrict__ t,
                        const float* __restrict__ xspec, float* __restrict__ spec2) {
  int idx = blockIdx.x * blockDim.x + threadIdx.x;
  if (idx >= kK * kD) return;
  int k = idx >> 6, d = idx & 63;
  float tc = fmaxf(t[d], 1e-8f);
  spec2[idx] = __expf(-evals[k] * tc) * xspec[idx];
}

// f = [x | x_diff | tanh(gX*b_re + gY*b_im)]  -> [NV, 192]
__global__ void k_gfeat(const float* __restrict__ x, const float* __restrict__ xd,
                        const float* __restrict__ gX, const float* __restrict__ gY,
                        const float* __restrict__ bre, const float* __restrict__ bim,
                        float* __restrict__ f) {
  int idx = blockIdx.x * blockDim.x + threadIdx.x;
  if (idx >= kNV * kD) return;
  int v = idx >> 6, d = idx & 63;
  float gf = tanhf(gX[idx] * bre[idx] + gY[idx] * bim[idx]);
  f[(size_t)v * 192 + d]       = x[idx];
  f[(size_t)v * 192 + 64 + d]  = xd[idx];
  f[(size_t)v * 192 + 128 + d] = gf;
}

__global__ void k_deg_init(float* __restrict__ deg) {
  int n = blockIdx.x * blockDim.x + threadIdx.x;
  if (n < kNG) deg[n] = 1.0f;  // self loop
}
__global__ void k_deg_edges(const int* __restrict__ dst, float* __restrict__ deg) {
  int e = blockIdx.x * blockDim.x + threadIdx.x;
  if (e < kE) atomicAdd(&deg[dst[e]], 1.0f);
}
__global__ void k_dinv(const float* __restrict__ deg, float* __restrict__ dinv) {
  int n = blockIdx.x * blockDim.x + threadIdx.x;
  if (n < kNG) dinv[n] = rsqrtf(deg[n]);  // deg >= 1 always
}

// agg init with the self-loop contribution: agg[n] = hlin[n] * dinv[n]^2
__global__ void k_agg_init(const float* __restrict__ hlin, const float* __restrict__ dinv,
                           float* __restrict__ agg) {
  int idx = blockIdx.x * blockDim.x + threadIdx.x;
  if (idx >= kNG * kD) return;
  float di = dinv[idx >> 6];
  agg[idx] = hlin[idx] * di * di;
}
__global__ void k_agg_scatter(const int* __restrict__ src, const int* __restrict__ dst,
                              const float* __restrict__ hlin, const float* __restrict__ dinv,
                              float* __restrict__ agg) {
  int idx = blockIdx.x * blockDim.x + threadIdx.x;
  if (idx >= kE * kD) return;
  int e = idx >> 6, d = idx & 63;
  int s = src[e], t = dst[e];
  float nrm = dinv[s] * dinv[t];
  atomicAdd(&agg[(size_t)t * kD + d], hlin[(size_t)s * kD + d] * nrm);
}
__global__ void k_agg_final(const float* __restrict__ agg, const float* __restrict__ b,
                            float* __restrict__ out, int do_relu) {
  int idx = blockIdx.x * blockDim.x + threadIdx.x;
  if (idx >= kNG * kD) return;
  float v = agg[idx] + b[idx & 63];
  out[idx] = do_relu ? fmaxf(v, 0.0f) : v;
}

// out[v][c] = diff_x[v] . last_w[:,c] + last_b[c]
__global__ void k_out(const float* __restrict__ x, const float* __restrict__ w,
                      const float* __restrict__ b, float* __restrict__ out) {
  int idx = blockIdx.x * blockDim.x + threadIdx.x;
  if (idx >= kNV * kCOUT) return;
  int v = idx >> 3, c = idx & 7;
  float acc = b[c];
  #pragma unroll
  for (int d = 0; d < kD; ++d) acc += x[(size_t)v * kD + d] * w[d * kCOUT + c];
  out[idx] = acc;
}

// ---------------------------------------------------------------------------
// Host orchestration
// ---------------------------------------------------------------------------
static void gemm(hipStream_t s, const float* A, int lda, const float* B, int ldb,
                 float* C, const float* bias, const float* add,
                 int M, int K, float alpha, int acc, int relu) {
  gemm_wmma_n64<<<dim3(M / 64), dim3(128), 0, s>>>(A, lda, B, ldb, C, bias, add,
                                                   M, K, alpha, acc, relu);
}

static float* wsf(void* ws, size_t& off, size_t bytes) {
  float* p = (float*)((char*)ws + off);
  off += (bytes + 255) & ~(size_t)255;
  return p;
}

extern "C" void kernel_launch(void* const* d_in, const int* in_sizes, int n_in,
                              void* d_out, int out_size, void* d_ws, size_t ws_size,
                              hipStream_t stream) {
  const float* surf_x    = (const float*)d_in[0];
  const float* mass      = (const float*)d_in[1];
  const float* evals     = (const float*)d_in[2];
  const float* evecs     = (const float*)d_in[3];
  const float* gradX     = (const float*)d_in[4];
  const float* gradY     = (const float*)d_in[5];
  const float* vertices  = (const float*)d_in[6];
  // d_in[7] graph_x, d_in[11] lin2_w, d_in[12] lin2_b: dead (gx is
  // overwritten by rbf.T @ diff_x before any use in the reference loop).
  const float* graph_pos = (const float*)d_in[8];
  const float* lin1_w    = (const float*)d_in[9];
  const float* lin1_b    = (const float*)d_in[10];
  const float* last_w    = (const float*)d_in[13];
  const float* last_b    = (const float*)d_in[14];
  const float* diff_time = (const float*)d_in[15];
  const float* A_re      = (const float*)d_in[16];
  const float* A_im      = (const float*)d_in[17];
  const float* mlp_w0    = (const float*)d_in[18];
  const float* mlp_b0    = (const float*)d_in[19];
  const float* mlp_w1    = (const float*)d_in[20];
  const float* mlp_b1    = (const float*)d_in[21];
  const float* mlp_w2    = (const float*)d_in[22];
  const float* mlp_b2    = (const float*)d_in[23];
  const float* gcn_w1    = (const float*)d_in[24];
  const float* gcn_b1    = (const float*)d_in[25];
  const float* gcn_w2    = (const float*)d_in[26];
  const float* gcn_b2    = (const float*)d_in[27];
  const int*   e_src     = (const int*)d_in[28];
  const int*   e_dst     = e_src + kE;
  float* out = (float*)d_out;

  // Workspace (~83 MB)
  size_t off = 0;
  float* rbf    = wsf(d_ws, off, (size_t)kNV * kNG * 4);
  float* rbfT   = wsf(d_ws, off, (size_t)kNV * kNG * 4);
  float* diff_x = wsf(d_ws, off, (size_t)kNV * kD * 4);
  float* xm     = wsf(d_ws, off, (size_t)kNV * kD * 4);
  float* xspec  = wsf(d_ws, off, (size_t)kK * kD * 4);
  float* spec2  = wsf(d_ws, off, (size_t)kK * kD * 4);
  float* x_diff = wsf(d_ws, off, (size_t)kNV * kD * 4);
  float* gX     = wsf(d_ws, off, (size_t)kNV * kD * 4);
  float* gY     = wsf(d_ws, off, (size_t)kNV * kD * 4);
  float* b_re   = wsf(d_ws, off, (size_t)kNV * kD * 4);
  float* b_im   = wsf(d_ws, off, (size_t)kNV * kD * 4);
  float* fcat   = wsf(d_ws, off, (size_t)kNV * 192 * 4);
  float* h1     = wsf(d_ws, off, (size_t)kNV * kD * 4);
  float* h2     = wsf(d_ws, off, (size_t)kNV * kD * 4);
  float* gx     = wsf(d_ws, off, (size_t)kNG * kD * 4);
  float* hlin   = wsf(d_ws, off, (size_t)kNG * kD * 4);
  float* agg    = wsf(d_ws, off, (size_t)kNG * kD * 4);
  float* hgr    = wsf(d_ws, off, (size_t)kNG * kD * 4);
  float* deg    = wsf(d_ws, off, (size_t)kNG * 4);
  float* dinv   = wsf(d_ws, off, (size_t)kNG * 4);
  (void)ws_size; (void)n_in; (void)in_sizes; (void)out_size;

  const int T = 256;

  // One-time precompute
  k_rbf<<<cdiv(kNV * kNG, T), T, 0, stream>>>(vertices, graph_pos, rbf, rbfT);
  k_lin1<<<cdiv(kNV * kD, T), T, 0, stream>>>(surf_x, lin1_w, lin1_b, diff_x);
  k_deg_init<<<cdiv(kNG, T), T, 0, stream>>>(deg);
  k_deg_edges<<<cdiv(kE, T), T, 0, stream>>>(e_dst, deg);
  k_dinv<<<cdiv(kNG, T), T, 0, stream>>>(deg, dinv);

  for (int i = 0; i < kNB; ++i) {
    const float* are = A_re + (size_t)i * kD * kD;
    const float* aim = A_im + (size_t)i * kD * kD;
    const float* w0  = mlp_w0 + (size_t)i * 3 * kD * kD;
    const float* b0  = mlp_b0 + (size_t)i * kD;
    const float* w1  = mlp_w1 + (size_t)i * kD * kD;
    const float* b1  = mlp_b1 + (size_t)i * kD;
    const float* w2  = mlp_w2 + (size_t)i * kD * kD;
    const float* b2  = mlp_b2 + (size_t)i * kD;
    const float* gw1 = gcn_w1 + (size_t)i * kD * kD;
    const float* gb1 = gcn_b1 + (size_t)i * kD;
    const float* gw2 = gcn_w2 + (size_t)i * kD * kD;
    const float* gb2 = gcn_b2 + (size_t)i * kD;
    const float* dt  = diff_time + (size_t)i * kD;

    // Spectral diffusion
    k_xm<<<cdiv(kNV * kD, T), T, 0, stream>>>(diff_x, mass, xm);
    k_spec<<<dim3(kK), dim3(kD), 0, stream>>>(evecs, xm, xspec);
    k_spec2<<<cdiv(kK * kD, T), T, 0, stream>>>(evals, dt, xspec, spec2);
    gemm(stream, evecs, kK, spec2, kD, x_diff, nullptr, nullptr, kNV, kK, 1.f, 0, 0);

    // Gradient features (the two big 4096x4096x64 GEMMs)
    gemm(stream, gradX, kNV, x_diff, kD, gX, nullptr, nullptr, kNV, kNV, 1.f, 0, 0);
    gemm(stream, gradY, kNV, x_diff, kD, gY, nullptr, nullptr, kNV, kNV, 1.f, 0, 0);

    // b_re = gX@A_re - gY@A_im ; b_im = gY@A_re + gX@A_im
    gemm(stream, gX, kD, are, kD, b_re, nullptr, nullptr, kNV, kD, 1.f, 0, 0);
    gemm(stream, gY, kD, aim, kD, b_re, nullptr, nullptr, kNV, kD, -1.f, 1, 0);
    gemm(stream, gY, kD, are, kD, b_im, nullptr, nullptr, kNV, kD, 1.f, 0, 0);
    gemm(stream, gX, kD, aim, kD, b_im, nullptr, nullptr, kNV, kD, 1.f, 1, 0);

    // MLP with residual
    k_gfeat<<<cdiv(kNV * kD, T), T, 0, stream>>>(diff_x, x_diff, gX, gY, b_re, b_im, fcat);
    gemm(stream, fcat, 192, w0, kD, h1, b0, nullptr, kNV, 192, 1.f, 0, 1);
    gemm(stream, h1, kD, w1, kD, h2, b1, nullptr, kNV, kD, 1.f, 0, 1);
    gemm(stream, h2, kD, w2, kD, diff_x, b2, diff_x, kNV, kD, 1.f, 0, 0);

    // surface -> graph
    gemm(stream, rbfT, kNV, diff_x, kD, gx, nullptr, nullptr, kNG, kNV, 1.f, 0, 0);

    // GCN conv1 (+relu)
    gemm(stream, gx, kD, gw1, kD, hlin, nullptr, nullptr, kNG, kD, 1.f, 0, 0);
    k_agg_init<<<cdiv(kNG * kD, T), T, 0, stream>>>(hlin, dinv, agg);
    k_agg_scatter<<<cdiv(kE * kD, T), T, 0, stream>>>(e_src, e_dst, hlin, dinv, agg);
    k_agg_final<<<cdiv(kNG * kD, T), T, 0, stream>>>(agg, gb1, hgr, 1);

    // GCN conv2
    gemm(stream, hgr, kD, gw2, kD, hlin, nullptr, nullptr, kNG, kD, 1.f, 0, 0);
    k_agg_init<<<cdiv(kNG * kD, T), T, 0, stream>>>(hlin, dinv, agg);
    k_agg_scatter<<<cdiv(kE * kD, T), T, 0, stream>>>(e_src, e_dst, hlin, dinv, agg);
    k_agg_final<<<cdiv(kNG * kD, T), T, 0, stream>>>(agg, gb2, gx, 0);

    // graph -> surface
    gemm(stream, rbf, kNG, gx, kD, diff_x, nullptr, nullptr, kNV, kNG, 1.f, 0, 0);
  }

  k_out<<<cdiv(kNV * kCOUT, T), T, 0, stream>>>(diff_x, last_w, last_b, out);
}
